// HeteroGNN_37460704756549
// MI455X (gfx1250) — matrix operand order repaired and checked by reading, not compiled, not run on hardware
//
#include <hip/hip_runtime.h>

#define F 256
#define KTOT 1024

typedef __attribute__((ext_vector_type(16))) __bf16 v16bf;
typedef __attribute__((ext_vector_type(8)))  float  v8f;

__device__ __forceinline__ unsigned short f2bf(float f) {
    unsigned int u = __float_as_uint(f);
    unsigned int r = u + 0x7FFFu + ((u >> 16) & 1u);   // round-to-nearest-even
    return (unsigned short)(r >> 16);
}
__device__ __forceinline__ float bf2f(unsigned short u) {
    return __uint_as_float(((unsigned int)u) << 16);
}

// ---------------------------------------------------------------------------
// Prep: Wt[n][k] bf16 stacked transposed weights (k segs: app, non, obs, root_sum),
// Wc = W_mlp @ W_lin (256x2), bsum = sum of rel biases, bc = fused classifier bias.
// One block of 256 threads.
// ---------------------------------------------------------------------------
__global__ void prep_kernel(const float* __restrict__ Wapp, const float* __restrict__ Wnon,
                            const float* __restrict__ Wobs,
                            const float* __restrict__ Wra,  const float* __restrict__ Wrn,
                            const float* __restrict__ Wro,
                            const float* __restrict__ Wmlp, const float* __restrict__ Wlin,
                            const float* __restrict__ ba,   const float* __restrict__ bn,
                            const float* __restrict__ bo,
                            const float* __restrict__ bmlp, const float* __restrict__ blin,
                            unsigned short* __restrict__ Wt, float* __restrict__ Wc,
                            float* __restrict__ bsum, float* __restrict__ bc)
{
    int n = threadIdx.x;                      // output-feature index 0..255
    for (int k = 0; k < F; ++k) {
        Wt[(size_t)n * KTOT + 0 * F + k] = f2bf(Wapp[k * F + n]);
        Wt[(size_t)n * KTOT + 1 * F + k] = f2bf(Wnon[k * F + n]);
        Wt[(size_t)n * KTOT + 2 * F + k] = f2bf(Wobs[k * F + n]);
        Wt[(size_t)n * KTOT + 3 * F + k] = f2bf(Wra[k * F + n] + Wrn[k * F + n] + Wro[k * F + n]);
    }
    float c0 = 0.f, c1 = 0.f;
    for (int j = 0; j < F; ++j) {
        float wm = Wmlp[n * F + j];
        c0 += wm * Wlin[j * 2 + 0];
        c1 += wm * Wlin[j * 2 + 1];
    }
    Wc[n * 2 + 0] = c0;
    Wc[n * 2 + 1] = c1;
    bsum[n] = ba[n] + bn[n] + bo[n];
    if (n < 2) {
        float acc = blin[n];
        for (int j = 0; j < F; ++j) acc += bmlp[j] * Wlin[j * 2 + n];
        bc[n] = acc;
    }
}

// ---------------------------------------------------------------------------
// Zero the fp32 aggregation buffers (float4 grid-stride).
// ---------------------------------------------------------------------------
__global__ void zero_kernel(float4* __restrict__ p, size_t n4)
{
    size_t i = (size_t)blockIdx.x * blockDim.x + threadIdx.x;
    size_t stride = (size_t)gridDim.x * blockDim.x;
    float4 z = make_float4(0.f, 0.f, 0.f, 0.f);
    for (; i < n4; i += stride) p[i] = z;
}

// ---------------------------------------------------------------------------
// Edge aggregation: one wave32 per edge; each lane moves 8 floats (2x float4
// gather, 8 global fp32 atomic adds into agg[dst]).  blockIdx.y = relation.
// ---------------------------------------------------------------------------
__global__ __launch_bounds__(256) void agg_kernel(
    const float* __restrict__ xa, const float* __restrict__ xn, const float* __restrict__ xo,
    const int* __restrict__ sa, const int* __restrict__ da,
    const int* __restrict__ sn, const int* __restrict__ dn,
    const int* __restrict__ so, const int* __restrict__ dk,
    float* __restrict__ agg, int E, int N)
{
    int rel = blockIdx.y;
    const float* x  = (rel == 0) ? xa : (rel == 1) ? xn : xo;
    const int* src  = (rel == 0) ? sa : (rel == 1) ? sn : so;
    const int* dst  = (rel == 0) ? da : (rel == 1) ? dn : dk;
    float* a = agg + (size_t)rel * (size_t)N * F;

    int wave = (int)(blockIdx.x * (blockDim.x >> 5)) + (threadIdx.x >> 5);
    int lane = threadIdx.x & 31;
    if (wave >= E) return;

    int s = src[wave];
    int d = dst[wave];
    const float4* xs = (const float4*)(x + (size_t)s * F);
    float4 v0 = xs[lane];          // floats [lane*4, lane*4+3]
    float4 v1 = xs[lane + 32];     // floats [128 + lane*4, ...]
    float* ad = a + (size_t)d * F;
    int o = lane * 4;
    atomicAdd(ad + o + 0,       v0.x);
    atomicAdd(ad + o + 1,       v0.y);
    atomicAdd(ad + o + 2,       v0.z);
    atomicAdd(ad + o + 3,       v0.w);
    atomicAdd(ad + 128 + o + 0, v1.x);
    atomicAdd(ad + 128 + o + 1, v1.y);
    atomicAdd(ad + 128 + o + 2, v1.z);
    atomicAdd(ad + 128 + o + 3, v1.w);
}

// ---------------------------------------------------------------------------
// Convert the fp32 A operand (3 agg buffers ++ x_tgt) into one contiguous
// bf16 matrix Abf[4*NF].  8 elements per thread per step (2x float4 in,
// 1x uint4 of packed bf16 out).
// ---------------------------------------------------------------------------
__global__ __launch_bounds__(256) void cvt_kernel(
    const float* __restrict__ agg, const float* __restrict__ xtgt,
    unsigned short* __restrict__ Abf, size_t NF)
{
    size_t n8 = (4 * NF) / 8;
    size_t i = (size_t)blockIdx.x * blockDim.x + threadIdx.x;
    size_t stride = (size_t)gridDim.x * blockDim.x;
    for (; i < n8; i += stride) {
        size_t e = i * 8;
        const float* src = (e < 3 * NF) ? (agg + e) : (xtgt + (e - 3 * NF));
        float4 f0 = *(const float4*)(src);
        float4 f1 = *(const float4*)(src + 4);
        union { unsigned short u[8]; uint4 v; } pk;
        pk.u[0] = f2bf(f0.x); pk.u[1] = f2bf(f0.y);
        pk.u[2] = f2bf(f0.z); pk.u[3] = f2bf(f0.w);
        pk.u[4] = f2bf(f1.x); pk.u[5] = f2bf(f1.y);
        pk.u[6] = f2bf(f1.z); pk.u[7] = f2bf(f1.w);
        *(uint4*)(Abf + e) = pk.v;
    }
}

// ---------------------------------------------------------------------------
// Fused GEMM:  z[64xF] = Abf(4 segs) @ Wt  (+bias, relu)
// then out[64x2] = relu(z) @ Wc + bc.
// Block: 256 threads = 8 waves; wave w owns columns [w*32, w*32+32), rows 0..63
// of the block tile as 4x2 WMMA 16x16 f32 accumulators.  All operands are
// bf16 row-major, each lane reading 16 contiguous K (k = 16*hi + s) for A and B -> inner loop is pure
// global_load_b128 + v_wmma.
// ---------------------------------------------------------------------------
__global__ __launch_bounds__(256) void gemm_kernel(
    const unsigned short* __restrict__ Abf,
    const unsigned short* __restrict__ Wt, const float* __restrict__ Wc,
    const float* __restrict__ bsum, const float* __restrict__ bc,
    float* __restrict__ out, int N)
{
    __shared__ unsigned short zs[64][F + 8];   // relu(z) staged as bf16

    const size_t NF = (size_t)N * F;
    int tid  = threadIdx.x;
    int wv   = tid >> 5;
    int lane = tid & 31;
    int lm   = lane & 15;     // m-in-tile (A/C) or n-in-tile (B/C)
    int hi   = lane >> 4;     // half-wave select
    int rowBase = blockIdx.x * 64;
    int n0 = wv * 32;

    v8f acc[4][2];
#pragma unroll
    for (int i = 0; i < 4; ++i)
#pragma unroll
        for (int j = 0; j < 2; ++j) acc[i][j] = (v8f)(0.0f);

    // precompute clamped row addresses (constant over K loop)
    const unsigned short* arow[4];
#pragma unroll
    for (int mt = 0; mt < 4; ++mt) {
        int m = rowBase + mt * 16 + lm;
        if (m >= N) m = N - 1;              // clamp (stores are guarded)
        arow[mt] = Abf + (size_t)m * F;
    }
    const unsigned short* brow[2];
#pragma unroll
    for (int c = 0; c < 2; ++c)
        brow[c] = Wt + (size_t)(n0 + c * 16 + lm) * KTOT;

#pragma unroll
    for (int seg = 0; seg < 4; ++seg) {
        size_t asegOff = (size_t)seg * NF;
        int bsegOff = seg * F;
        for (int kt = 0; kt < 8; ++kt) {
            // ---- B fragments: per-lane K contiguous (lanes 0-15 -> K 0..15,
            //      lanes 16-31 -> K 16..31 of this 32-step)
            v16bf bfrag[2];
#pragma unroll
            for (int c = 0; c < 2; ++c) {
                const unsigned short* bp = brow[c] + bsegOff + kt * 32 + hi * 16;
                union { uint4 u[2]; v16bf v; } ub;
                ub.u[0] = *(const uint4*)(bp);
                ub.u[1] = *(const uint4*)(bp + 8);
                bfrag[c] = ub.v;
            }
            // ---- A fragments: lanes<16 hold K 0..15, lanes>=16 hold
            //      K 16..31 (contiguous, matching B) -> two b128 loads
            int kA = kt * 32 + hi * 16;
#pragma unroll
            for (int mt = 0; mt < 4; ++mt) {
                const unsigned short* ap = arow[mt] + asegOff + kA;
                union { uint4 u[2]; v16bf v; } ua;
                ua.u[0] = *(const uint4*)(ap);
                ua.u[1] = *(const uint4*)(ap + 8);
#pragma unroll
                for (int c = 0; c < 2; ++c) {
                    acc[mt][c] = __builtin_amdgcn_wmma_f32_16x16x32_bf16(
                        false, ua.v, false, bfrag[c],
                        (short)0, acc[mt][c], false, false);
                }
            }
        }
    }

    // ---- epilogue: bias + relu -> LDS (bf16).  C layout: VGPR r holds
    //      m = r + 8*hi (lanes mod 16 give n).
#pragma unroll
    for (int mt = 0; mt < 4; ++mt) {
#pragma unroll
        for (int c = 0; c < 2; ++c) {
            int n = n0 + c * 16 + lm;
            float bias = bsum[n];
#pragma unroll
            for (int r = 0; r < 8; ++r) {
                int ml = mt * 16 + r + 8 * hi;
                float v = acc[mt][c][r] + bias;
                zs[ml][n] = f2bf(v > 0.f ? v : 0.f);
            }
        }
    }
    __syncthreads();

    // ---- 256 -> 2 projection with fused classifier weight
    if (tid < 128) {
        int row = tid >> 1;
        int o   = tid & 1;
        float a = bc[o];
        for (int n = 0; n < F; ++n) a += bf2f(zs[row][n]) * Wc[n * 2 + o];
        int gr = rowBase + row;
        if (gr < N) out[(size_t)gr * 2 + o] = a;
    }
}

// ---------------------------------------------------------------------------
extern "C" void kernel_launch(void* const* d_in, const int* in_sizes, int n_in,
                              void* d_out, int out_size, void* d_ws, size_t ws_size,
                              hipStream_t stream)
{
    const float* x_app = (const float*)d_in[0];
    const float* x_non = (const float*)d_in[1];
    const float* x_obs = (const float*)d_in[2];
    const float* x_tgt = (const float*)d_in[3];
    const int* src_app = (const int*)d_in[4];
    const int* dst_app = (const int*)d_in[5];
    const int* src_non = (const int*)d_in[6];
    const int* dst_non = (const int*)d_in[7];
    const int* src_obs = (const int*)d_in[8];
    const int* dst_obs = (const int*)d_in[9];
    const float* W_rel_app  = (const float*)d_in[10];
    const float* b_rel_app  = (const float*)d_in[11];
    const float* W_root_app = (const float*)d_in[12];
    const float* W_rel_non  = (const float*)d_in[13];
    const float* b_rel_non  = (const float*)d_in[14];
    const float* W_root_non = (const float*)d_in[15];
    const float* W_rel_obs  = (const float*)d_in[16];
    const float* b_rel_obs  = (const float*)d_in[17];
    const float* W_root_obs = (const float*)d_in[18];
    const float* W_mlp = (const float*)d_in[19];
    const float* b_mlp = (const float*)d_in[20];
    const float* W_lin = (const float*)d_in[21];
    const float* b_lin = (const float*)d_in[22];

    int N = in_sizes[0] / F;
    int E = in_sizes[4];
    size_t NF = (size_t)N * F;

    // workspace layout
    float* agg           = (float*)d_ws;                       // 3*NF f32
    unsigned short* Abf  = (unsigned short*)(agg + 3 * NF);    // 4*NF bf16
    unsigned short* Wt   = Abf + 4 * NF;                       // F*KTOT bf16
    float* Wc            = (float*)(Wt + (size_t)F * KTOT);    // F*2 f32
    float* bsum          = Wc + 2 * F;                         // F
    float* bc            = bsum + F;                           // 2
    float* out           = (float*)d_out;

    prep_kernel<<<1, 256, 0, stream>>>(
        W_rel_app, W_rel_non, W_rel_obs,
        W_root_app, W_root_non, W_root_obs,
        W_mlp, W_lin,
        b_rel_app, b_rel_non, b_rel_obs,
        b_mlp, b_lin,
        Wt, Wc, bsum, bc);

    zero_kernel<<<2048, 256, 0, stream>>>((float4*)agg, (3 * NF) / 4);

    dim3 agrid((E + 7) / 8, 3);
    agg_kernel<<<agrid, 256, 0, stream>>>(
        x_app, x_non, x_obs,
        src_app, dst_app, src_non, dst_non, src_obs, dst_obs,
        agg, E, N);

    cvt_kernel<<<4096, 256, 0, stream>>>(agg, x_tgt, Abf, NF);

    gemm_kernel<<<(N + 63) / 64, 256, 0, stream>>>(
        Abf, Wt, Wc, bsum, bc, out, N);
}
